// ProductGumbelVQ_65953517797735
// MI455X (gfx1250) — matrix operand order repaired and checked by reading, not compiled, not run on hardware
//
#include <hip/hip_runtime.h>

// ---------------------------------------------------------------------------
// Product-VQ forward for MI455X (gfx1250, wave32, WMMA).
// GEMM on v_wmma_f32_16x16x32_bf16. Fused argmax/softmax/probs so logits
// (512MB fp32) never hit HBM.  HBM ~= 512MB (z read + z_q write) ~= 22us.
// B-fragments reused across 4 row tiles per load -> codebook L2 traffic 1GB.
// ---------------------------------------------------------------------------

typedef __attribute__((ext_vector_type(16))) __bf16 v16bf;
typedef __attribute__((ext_vector_type(8)))  float  v8f;

#define B_TOT   65536
#define H_N     4
#define K_N     512
#define DH      256
#define ZROWF   1024            // H*DH floats per z_e row
#define WAVES   4
#define BROWS   64              // rows per block
#define RT_N    4               // 16-row tiles per block
#define CT_W    8               // column tiles per wave (8*16 = 128 codewords)
#define ZBF_S   264             // bf16 z tile stride   (bank-conflict pad)
#define EXP_S   520             // bf16 exp tile stride (row = 1040 bytes)

union AFrag { v16bf v; uint4 q[2]; };

// --------------------------- codebook prep (emb -> normalized bf16) --------
__global__ __launch_bounds__(128)
void vq_prep(const float* __restrict__ emb, __bf16* __restrict__ en16) {
  const int row  = blockIdx.x * 4 + (threadIdx.x >> 5);   // h*K + k
  const int lane = threadIdx.x & 31;
  const float4* src = (const float4*)(emb + (size_t)row * DH);
  float4 a0 = src[lane];
  float4 a1 = src[lane + 32];
  float ss = a0.x*a0.x + a0.y*a0.y + a0.z*a0.z + a0.w*a0.w
           + a1.x*a1.x + a1.y*a1.y + a1.z*a1.z + a1.w*a1.w;
#pragma unroll
  for (int d = 1; d < 32; d <<= 1) ss += __shfl_xor(ss, d, 32);
  const float rn = 1.0f / fmaxf(sqrtf(ss), 1e-12f);
  union { __bf16 h[4]; uint2 u; } p0, p1;
  p0.h[0]=(__bf16)(a0.x*rn); p0.h[1]=(__bf16)(a0.y*rn);
  p0.h[2]=(__bf16)(a0.z*rn); p0.h[3]=(__bf16)(a0.w*rn);
  p1.h[0]=(__bf16)(a1.x*rn); p1.h[1]=(__bf16)(a1.y*rn);
  p1.h[2]=(__bf16)(a1.z*rn); p1.h[3]=(__bf16)(a1.w*rn);
  uint2* dst = (uint2*)(en16 + (size_t)row * DH);
  dst[lane]      = p0.u;
  dst[lane + 32] = p1.u;
}

// --------------------------- zero probs accumulator ------------------------
__global__ void vq_zero(float* __restrict__ acc) {
  acc[blockIdx.x * 256 + threadIdx.x] = 0.f;
}

// --------------------------- main fused kernel -----------------------------
__global__ __launch_bounds__(128)
void vq_main(const float* __restrict__ z_e, const float* __restrict__ emb,
             const float* __restrict__ scales, const __bf16* __restrict__ en16,
             float* __restrict__ zq, int* __restrict__ comb,
             float* __restrict__ acc) {
  extern __shared__ char smem[];
  // exp tile [64][EXP_S] bf16; rows are 1040B -> also used as f32 z staging
  __bf16* expb = (__bf16*)smem;
  __bf16* zbf  = (__bf16*)(smem + BROWS * EXP_S * 2);          // [64][ZBF_S]
  float*  partM = (float*)(smem + BROWS * EXP_S * 2 + BROWS * ZBF_S * 2);
  float*  partS = partM + WAVES * BROWS;
  int*    partI = (int*)(partS + WAVES * BROWS);
  float*  rsArr = (float*)(partI + WAVES * BROWS);             // [64] 1/rowSum
  int*    idxArr = (int*)(rsArr + BROWS);                      // [H_N][64]

  const int tid  = threadIdx.x;
  const int wave = tid >> 5;
  const int lane = tid & 31;
  const int hi   = lane >> 4;
  const int lm   = lane & 15;
  const int row0 = blockIdx.x * BROWS;

  for (int h = 0; h < H_N; ++h) {
    // ---- stage: wave w loads its 16 rows (f32) into exp region ----
    const float* zsrc = z_e + (size_t)(row0 + wave * 16) * ZROWF + h * DH;
#pragma unroll 4
    for (int r = 0; r < 16; ++r) {
      float* drow = (float*)((char*)expb + (size_t)(wave * 16 + r) * (EXP_S * 2));
      const float4* s4 = (const float4*)(zsrc + (size_t)r * ZROWF);
      ((float4*)drow)[lane]         = s4[lane];
      ((float4*)(drow + 128))[lane] = s4[lane + 32];
    }
    // ---- norm of row wave*16+lm (lanes L / L+16 split halves) ----
    const int myrow = wave * 16 + lm;
    float ss = 0.f;
    {
      const float4* zr = (const float4*)((char*)expb + (size_t)myrow * (EXP_S * 2)) + hi * 32;
#pragma unroll
      for (int i = 0; i < 32; ++i) {
        float4 x = zr[i];
        ss += x.x*x.x + x.y*x.y + x.z*x.z + x.w*x.w;
      }
    }
    ss += __shfl_xor(ss, 16, 32);
    const float rn = 1.0f / fmaxf(sqrtf(ss), 1e-12f);
    // ---- convert my half-row to bf16 z tile (A-fragment source) ----
    {
      const float* srow = (const float*)((char*)expb + (size_t)myrow * (EXP_S * 2)) + hi * 128;
      __bf16* drow = zbf + (size_t)myrow * ZBF_S + hi * 128;
#pragma unroll
      for (int i = 0; i < 16; ++i) {
        float4 x0 = ((const float4*)srow)[2 * i];
        float4 x1 = ((const float4*)srow)[2 * i + 1];
        union { __bf16 hh[8]; uint4 q; } p;
        p.hh[0]=(__bf16)(x0.x*rn); p.hh[1]=(__bf16)(x0.y*rn);
        p.hh[2]=(__bf16)(x0.z*rn); p.hh[3]=(__bf16)(x0.w*rn);
        p.hh[4]=(__bf16)(x1.x*rn); p.hh[5]=(__bf16)(x1.y*rn);
        p.hh[6]=(__bf16)(x1.z*rn); p.hh[7]=(__bf16)(x1.w*rn);
        ((uint4*)drow)[i] = p.q;
      }
    }
    __syncthreads();   // zbf complete; exp region free to overwrite

    // ---- GEMM: wave owns 8 column tiles, B frag reused over 4 row tiles ---
    const float scale = scales[h];
    float mA[RT_N][8], sA[RT_N][8]; int iA[RT_N][8];
#pragma unroll
    for (int rt = 0; rt < RT_N; ++rt)
#pragma unroll
      for (int j = 0; j < 8; ++j) { mA[rt][j] = -3.4e38f; sA[rt][j] = 0.f; iA[rt][j] = K_N; }

    const __bf16* eh = en16 + (size_t)h * K_N * DH;
    for (int i8 = 0; i8 < CT_W; ++i8) {
      const int c = (wave * CT_W + i8) * 16 + lm;        // this lane's codeword
      const __bf16* ecol = eh + (size_t)c * DH + hi * 16;
      v8f cf[RT_N];
#pragma unroll
      for (int rt = 0; rt < RT_N; ++rt) cf[rt] = (v8f){};
#pragma unroll
      for (int kk = 0; kk < 8; ++kk) {
        v16bf bv = *(const v16bf*)(ecol + kk * 32);      // one B frag / (ct,kk)
        const __bf16* ab = zbf + (size_t)lm * ZBF_S + kk * 32 + hi * 16;
#pragma unroll
        for (int rt = 0; rt < RT_N; ++rt) {              // 4 independent chains
          AFrag a;
          const __bf16* ar = ab + (size_t)rt * 16 * ZBF_S;
          a.q[0] = *(const uint4*)(ar);
          a.q[1] = *(const uint4*)(ar + 8);
          cf[rt] = __builtin_amdgcn_wmma_f32_16x16x32_bf16(
              false, a.v, false, bv, (short)0, cf[rt], false, false);
        }
      }
#pragma unroll
      for (int rt = 0; rt < RT_N; ++rt)
#pragma unroll
        for (int j = 0; j < 8; ++j) {                    // C: row rt*16+j+8*hi
          float v = cf[rt][j] * scale;
          float e = __expf(v);                           // logits <= ~10: safe
          sA[rt][j] += e;
          if (v > mA[rt][j]) { mA[rt][j] = v; iA[rt][j] = c; }
          expb[(size_t)(rt * 16 + j + 8 * hi) * EXP_S + c] = (__bf16)e;
        }
    }

    // ---- per-wave partial reductions over its 16 lane-columns ----
#pragma unroll
    for (int rt = 0; rt < RT_N; ++rt)
#pragma unroll
      for (int j = 0; j < 8; ++j) {
        float mm = mA[rt][j]; int ii = iA[rt][j]; float sm = sA[rt][j];
#pragma unroll
        for (int d = 1; d < 16; d <<= 1) {
          float om = __shfl_xor(mm, d, 32);
          int   oi = __shfl_xor(ii, d, 32);
          float os = __shfl_xor(sm, d, 32);
          sm += os;
          if (om > mm || (om == mm && oi < ii)) { mm = om; ii = oi; }
        }
        if (lm == 0) {
          const int r = rt * 16 + j + 8 * hi;
          partM[wave * BROWS + r] = mm;
          partS[wave * BROWS + r] = sm;
          partI[wave * BROWS + r] = ii;
        }
      }
    __syncthreads();
    // ---- combine 4 wave partials per row (waves = ordered col ranges) ----
    if (tid < BROWS) {
      float mm = partM[tid]; int ii = partI[tid]; float sm = partS[tid];
#pragma unroll
      for (int w = 1; w < WAVES; ++w) {
        float om = partM[w * BROWS + tid];
        int   oi = partI[w * BROWS + tid];
        sm += partS[w * BROWS + tid];
        if (om > mm) { mm = om; ii = oi; }   // strict >: keep lowest index
      }
      rsArr[tid] = 1.0f / sm;
      idxArr[h * BROWS + tid] = ii;
    }
    __syncthreads();
    // ---- probs: one column per thread-slot, one global atomic each ----
    {
      float t0 = 0.f, t1 = 0.f, t2 = 0.f, t3 = 0.f;
      for (int r = 0; r < BROWS; ++r) {
        const float rs = rsArr[r];
        const __bf16* er = expb + (size_t)r * EXP_S;
        t0 += (float)er[tid]       * rs;
        t1 += (float)er[tid + 128] * rs;
        t2 += (float)er[tid + 256] * rs;
        t3 += (float)er[tid + 384] * rs;
      }
      atomicAdd(&acc[h * K_N + tid],       t0);
      atomicAdd(&acc[h * K_N + tid + 128], t1);
      atomicAdd(&acc[h * K_N + tid + 256], t2);
      atomicAdd(&acc[h * K_N + tid + 384], t3);
    }
    __syncthreads();   // exp/rsArr free for next head
  }

  // ---- combined code (int32 wraparound matches JAX int32 semantics) ----
  if (tid < BROWS) {
    unsigned cm = (unsigned)idxArr[0 * BROWS + tid];
#pragma unroll
    for (int h = 1; h < H_N; ++h)
      cm = cm * (unsigned)K_N + (unsigned)idxArr[h * BROWS + tid];
    comb[row0 + tid] = (int)cm;
  }

  // ---- z_q gather from ORIGINAL emb (L2-resident), coalesced b128 ----
  {
    const int rbase = row0 + wave * 16;
    for (int r = 0; r < 16; ++r) {
      float4* drow = (float4*)(zq + (size_t)(rbase + r) * ZROWF);
#pragma unroll
      for (int h = 0; h < H_N; ++h) {
        const int id = idxArr[h * BROWS + wave * 16 + r];
        const float4* se = (const float4*)(emb + ((size_t)h * K_N + id) * DH);
        drow[h * 64 + lane]      = se[lane];
        drow[h * 64 + lane + 32] = se[lane + 32];
      }
    }
  }
}

// --------------------------- perplexity reduction --------------------------
__global__ __launch_bounds__(256)
void vq_perp(const float* __restrict__ acc, float* __restrict__ out) {
  __shared__ float red[256];
  const int tid = threadIdx.x;
  float result = 0.f;
  for (int h = 0; h < H_N; ++h) {
    float part = 0.f;
    for (int c = tid; c < K_N; c += 256) {
      float a = acc[h * K_N + c] * (1.0f / 65536.0f);   // mean over B
      part += a * __logf(a + 1e-10f);
    }
    red[tid] = part; __syncthreads();
    for (int s = 128; s > 0; s >>= 1) {
      if (tid < s) red[tid] += red[tid + s];
      __syncthreads();
    }
    if (tid == 0) result += __expf(-red[0]);
    __syncthreads();
  }
  if (tid == 0) out[0] = result * (1.0f / H_N);
}

// --------------------------- host launcher ---------------------------------
extern "C" void kernel_launch(void* const* d_in, const int* in_sizes, int n_in,
                              void* d_out, int out_size, void* d_ws, size_t ws_size,
                              hipStream_t stream) {
  (void)in_sizes; (void)n_in; (void)out_size; (void)ws_size;
  const float* z_e    = (const float*)d_in[0];
  const float* emb    = (const float*)d_in[1];
  const float* scales = (const float*)d_in[2];

  float* out  = (float*)d_out;
  float* zqp  = out;                                         // B x 1024 f32
  int*   comb = (int*)(out + (size_t)B_TOT * H_N * DH);      // B int32 (raw bits)
  float* perp = out + (size_t)B_TOT * H_N * DH + B_TOT;      // scalar

  __bf16* en16 = (__bf16*)d_ws;                              // 1 MB bf16 codebook
  float*  acc  = (float*)((char*)d_ws +
                          (size_t)H_N * K_N * DH * sizeof(__bf16)); // 8 KB

  vq_prep<<<(H_N * K_N) / 4, 128, 0, stream>>>(emb, en16);
  vq_zero<<<(H_N * K_N) / 256, 256, 0, stream>>>(acc);

  const size_t shmem =
      (size_t)BROWS * EXP_S * 2 +                 // exp tile / z staging
      (size_t)BROWS * ZBF_S * 2 +                 // bf16 z tile
      (size_t)WAVES * BROWS * 4 * 3 +             // partM/partS/partI
      (size_t)BROWS * 4 +                         // rsArr
      (size_t)H_N * BROWS * 4;                    // idxArr   (~105 KB total)
  vq_main<<<B_TOT / BROWS, 128, shmem, stream>>>(z_e, emb, scales, en16,
                                                 zqp, comb, acc);
  vq_perp<<<1, 256, 0, stream>>>(acc, perp);
}
